// MvCnnDownLayer_51797305590252
// MI455X (gfx1250) — matrix operand order repaired and checked by reading, not compiled, not run on hardware
//
#include <hip/hip_runtime.h>
#include <stdint.h>

// MI455X / gfx1250, wave32.
// Row-recurrent conv1d: per H-step GEMM (M=256, N=32, K=160) on bf16 WMMA,
// f32 accumulate. One barrier/step; carry kept in registers and fused into
// the next row's x-buffer build; input rows prefetched with async global->LDS.

typedef __attribute__((ext_vector_type(16))) __bf16 v16bf;
typedef __attribute__((ext_vector_type(8)))  float  v8f;

#define Bn 16
#define Hn 256
#define Wn 256
#define Cn 32
#define Kn 5

#define ROW_FLOATS   (Wn * Cn)                    // 8192
#define STAGE_BYTES  (2 * ROW_FLOATS * 4)         // 65536: double-buffered input row
#define XBUF_SHORTS  ((Wn + 4) * Cn)              // 8320 bf16 incl. 2-row halos
#define SMEM_BYTES   (STAGE_BYTES + 2 * XBUF_SHORTS * 2)   // ~96.5 KB

__device__ __forceinline__ unsigned short f2bf(float f) {
  unsigned u = __float_as_uint(f);
  u += 0x7FFFu + ((u >> 16) & 1u);   // round-to-nearest-even
  return (unsigned short)(u >> 16);
}

union Frag16 { v16bf v; unsigned short s[16]; uint4 u[2]; };

// Each wave async-loads exactly its own 512-float span of the row, so its own
// s_wait_asynccnt 0 guarantees the region it later reads has landed (ASYNCcnt
// is per-wave; no cross-wave dependence, no barrier needed for the stage).
__device__ __forceinline__ void async_row(const float* rp, unsigned lds_dst_base,
                                          int wv, int lane) {
  #pragma unroll
  for (int j = 0; j < 4; ++j) {
    int q = wv * 128 + lane + 32 * j;            // 16B chunks, wave-local
    unsigned           ldst = lds_dst_base + 16u * (unsigned)q;
    unsigned long long ga   = (unsigned long long)(uintptr_t)(rp + 4 * q);
    asm volatile("global_load_async_to_lds_b128 %0, %1, off"
                 :: "v"(ldst), "v"(ga) : "memory");
  }
}

__global__ __launch_bounds__(512) void mvcnn_down_kernel(
    const float* __restrict__ gin,   // [B,H,W,C]
    const float* __restrict__ gw,    // [C,K,1,C] -> idx c*160 + k*32 + o
    const float* __restrict__ gb,    // [C]
    float* __restrict__ gout) {      // [B,H,W,C]
  extern __shared__ char smem[];
  float*          stage = (float*)smem;                       // 2 x input row, f32
  unsigned short* xb0   = (unsigned short*)(smem + STAGE_BYTES);
  unsigned short* xb1   = xb0 + XBUF_SHORTS;

  const int t    = threadIdx.x;
  const int b    = blockIdx.x;
  const int lane = t & 31;
  const int wv   = t >> 5;       // 16 waves, each owns 16 W rows
  const int ml   = lane & 15;    // WMMA: n (B/C/D) and m-low (A)
  const int g    = lane >> 4;    // lane half-group
  const int w0   = wv * 16;

  // ---- preload B fragments (weights) in exact 16-bit B VGPR layout ----
  // lane: n = ml (+16 for half h), vector elem e -> input channel c = g*8 + e (e<8), 16 + g*8 + (e-8) (e>=8)
  v16bf Bf[Kn][2];
  #pragma unroll
  for (int k = 0; k < Kn; ++k) {
    #pragma unroll
    for (int h = 0; h < 2; ++h) {
      Frag16 f;
      #pragma unroll
      for (int e = 0; e < 16; ++e) {
        int c = (e < 8) ? (g * 8 + e) : (16 + g * 8 + (e - 8));
        int o = ml + h * 16;
        f.s[e] = f2bf(gw[c * (Kn * Cn) + k * Cn + o]);
      }
      Bf[k][h] = f.v;
    }
  }
  const float bias0 = gb[ml];
  const float bias1 = gb[ml + 16];

  const unsigned stage_lds = (unsigned)(uintptr_t)(void*)stage; // low32 flat = LDS off
  const size_t   in_base   = (size_t)b * Hn * ROW_FLOATS;

  // ---- prologue: stage row 0, zero halos, build x_0 = in_0 (carry = 0) ----
  async_row(gin + in_base, stage_lds, wv, lane);
  if (t < 2 * Cn) {                     // zero halo columns of both x buffers
    xb0[t] = 0; xb0[2 * Cn + Wn * Cn + t] = 0;
    xb1[t] = 0; xb1[2 * Cn + Wn * Cn + t] = 0;
  }
  asm volatile("s_wait_asynccnt 0x0" ::: "memory");
  #pragma unroll
  for (int j = 0; j < 8; ++j) {
    int wgl = w0 + g * 8 + j;
    xb0[(wgl + 2) * Cn + ml]      = f2bf(stage[wgl * Cn + ml]);
    xb0[(wgl + 2) * Cn + ml + 16] = f2bf(stage[wgl * Cn + ml + 16]);
  }
  __syncthreads();

  for (int i = 0; i < Hn; ++i) {
    // prefetch row i+1 while we compute row i
    if (i + 1 < Hn) {
      async_row(gin + in_base + (size_t)(i + 1) * ROW_FLOATS,
                stage_lds + (unsigned)(((i + 1) & 1) * (ROW_FLOATS * 4)),
                wv, lane);
    }

    // ---- GEMM tile 16x32: 10 x v_wmma_f32_16x16x32_bf16 over 5 taps ----
    const unsigned short* xb = (i & 1) ? xb1 : xb0;
    v8f acc0 = {};
    v8f acc1 = {};
    const int wrow = w0 + ml;    // A: m = ml; halo makes tap-k base = wrow + k
    #pragma unroll
    for (int k = 0; k < Kn; ++k) {
      // 16-bit A layout: elems 0-7 = ch g*8.., elems 8-15 = ch 16+g*8..
      Frag16 A;
      const unsigned short* ap = xb + (wrow + k) * Cn + g * 8;
      A.u[0] = *(const uint4*)ap;
      A.u[1] = *(const uint4*)(ap + 16);
      acc0 = __builtin_amdgcn_wmma_f32_16x16x32_bf16(
          false, A.v, false, Bf[k][0], (short)0, acc0, false, false);
      acc1 = __builtin_amdgcn_wmma_f32_16x16x32_bf16(
          false, A.v, false, Bf[k][1], (short)0, acc1, false, false);
    }

    // ---- epilogue: r = relu(acc + bias); fuse carry into next x buffer ----
    // C/D layout: lane holds n = ml (+16), rows m = g*8 + j for VGPR j
    float r0[8], r1[8];
    #pragma unroll
    for (int j = 0; j < 8; ++j) {
      r0[j] = fmaxf(acc0[j] + bias0, 0.f);
      r1[j] = fmaxf(acc1[j] + bias1, 0.f);
    }

    if (i + 1 < Hn) {
      asm volatile("s_wait_asynccnt 0x0" ::: "memory");  // our span of row i+1
      const float*    stN = stage + ((i + 1) & 1) * ROW_FLOATS;
      unsigned short* xbN = ((i + 1) & 1) ? xb1 : xb0;
      #pragma unroll
      for (int j = 0; j < 8; ++j) {
        int wgl = w0 + g * 8 + j;
        xbN[(wgl + 2) * Cn + ml]      = f2bf(stN[wgl * Cn + ml]      + r0[j]);
        xbN[(wgl + 2) * Cn + ml + 16] = f2bf(stN[wgl * Cn + ml + 16] + r1[j]);
      }
    }

    float* orow = gout + in_base + (size_t)i * ROW_FLOATS;
    #pragma unroll
    for (int j = 0; j < 8; ++j) {
      int wgl = w0 + g * 8 + j;
      orow[wgl * Cn + ml]      = r0[j];
      orow[wgl * Cn + ml + 16] = r1[j];
    }

    __syncthreads();   // halo columns of xbN visible to neighbor waves
  }
}

extern "C" void kernel_launch(void* const* d_in, const int* in_sizes, int n_in,
                              void* d_out, int out_size, void* d_ws, size_t ws_size,
                              hipStream_t stream) {
  (void)in_sizes; (void)n_in; (void)out_size; (void)d_ws; (void)ws_size;
  const float* gin = (const float*)d_in[0];   // [16,256,256,32] f32
  const float* gw  = (const float*)d_in[1];   // [32,5,1,32] f32
  const float* gb  = (const float*)d_in[2];   // [32] f32
  float* gout = (float*)d_out;                // [16,256,256,32] f32
  mvcnn_down_kernel<<<dim3(Bn), dim3(512), SMEM_BYTES, stream>>>(gin, gw, gb, gout);
}
